// JointAttention_7928509629171
// MI455X (gfx1250) — matrix operand order repaired and checked by reading, not compiled, not run on hardware
//
#include <hip/hip_runtime.h>
#include <hip/hip_bf16.h>
#include <cstdint>

// ---------------------------------------------------------------------------
// MI455X (gfx1250, wave32) joint-attention pipeline, bf16 WMMA everywhere.
// Fragment loads are b128 (layout-exploiting), K/Q attention tiles staged via
// global_load_async_to_lds_b128 (ASYNCcnt path).
// ---------------------------------------------------------------------------

typedef __bf16 bf16_t;
typedef __attribute__((ext_vector_type(16))) __bf16 v16bf;
typedef __attribute__((ext_vector_type(8)))  float  v8f;

#define BATCH     4
#define SEQ       2048
#define DIM       1024
#define HEADS     16
#define HEAD_DIM  64
#define KV_TOTAL  2880   // 2048 + 512 + 64 + 256

union Frag16 {           // one 16x32 bf16 WMMA operand = two b128 LDS loads
    uint4 q[2];
    v16bf v;
};
union H8 { uint4 q; bf16_t h[8]; };
union H4 { uint2 u; bf16_t h[4]; };

static __device__ inline v8f v8f_zero() {
    v8f z = {0.f, 0.f, 0.f, 0.f, 0.f, 0.f, 0.f, 0.f};
    return z;
}

// Async global -> LDS, 16 bytes per lane (ASYNCcnt tracked).
#define ASYNC_G2L_B128(ldsoff, gptr)                                   \
    asm volatile("global_load_async_to_lds_b128 %0, %1, off"           \
                 :: "v"(ldsoff), "v"(gptr) : "memory")
#define WAIT_ASYNCCNT0()                                               \
    asm volatile("s_wait_asynccnt 0" ::: "memory")

// ---------------------------------------------------------------------------
// Generic GEMM: C[M,N] = A[M,K] * W[N,K]^T    (W is the (out,in) weight)
// Block tile 128x64, 8 waves each owning a 32x32 quadrant (2x2 fragments),
// K step 32 via v_wmma_f32_16x16x32_bf16.
//   EPI==0 : f32   C[(b*obs + oro + rib)*N + col]
//   EPI==1 : bf16  scatter to [b, h=col/64, kv=oro+rib, d=col%64]
// M multiple of 128, N of 64, K of 32; rows_per_batch = 1<<rpb_shift.
// ---------------------------------------------------------------------------
template <typename AT, int EPI>
__global__ __launch_bounds__(256)
void gemm_wmma_kernel(const AT* __restrict__ A, const float* __restrict__ W,
                      void* __restrict__ Cout,
                      int M, int N, int K,
                      int rpb_shift, int out_batch_stride,
                      int out_row_offset, int kv_total)
{
    __shared__ bf16_t As[128][40];   // 80B row stride: 16B aligned, no 16-lane conflicts
    __shared__ bf16_t Bs[64][40];

    const int tid  = threadIdx.x;
    const int lane = tid & 31;
    const int wave = tid >> 5;
    const int wm   = wave >> 1;          // 0..3  (M quadrant)
    const int wn   = wave & 1;           // 0..1  (N quadrant)
    const int bm   = blockIdx.x * 128;
    const int bn   = blockIdx.y * 64;
    const int half = lane >> 4;
    const int l16  = lane & 15;

    v8f acc[2][2];
    acc[0][0] = v8f_zero(); acc[0][1] = v8f_zero();
    acc[1][0] = v8f_zero(); acc[1][1] = v8f_zero();

    for (int k0 = 0; k0 < K; k0 += 32) {
        // ---- stage A tile (128x32) as bf16, b128 global loads ----
        if constexpr (sizeof(AT) == 2) {
            #pragma unroll
            for (int i = 0; i < 2; ++i) {
                int e = tid + i * 256;          // 8-elem chunk id (512 total)
                int r = e >> 2, c8 = (e & 3) * 8;
                *(uint4*)&As[r][c8] =
                    *(const uint4*)&A[(size_t)(bm + r) * K + k0 + c8];
            }
        } else {
            #pragma unroll
            for (int i = 0; i < 4; ++i) {
                int e = tid + i * 256;          // float4 chunk id (1024 total)
                int r = e >> 3, c4 = (e & 7) * 4;
                float4 f = *(const float4*)&A[(size_t)(bm + r) * K + k0 + c4];
                H4 t; t.h[0] = (bf16_t)f.x; t.h[1] = (bf16_t)f.y;
                      t.h[2] = (bf16_t)f.z; t.h[3] = (bf16_t)f.w;
                *(uint2*)&As[r][c4] = t.u;
            }
        }
        // ---- stage W tile (64x32) as bf16 ----
        #pragma unroll
        for (int i = 0; i < 2; ++i) {
            int e = tid + i * 256;              // float4 chunk id (512 total)
            int r = e >> 3, c4 = (e & 7) * 4;
            float4 f = *(const float4*)&W[(size_t)(bn + r) * K + k0 + c4];
            H4 t; t.h[0] = (bf16_t)f.x; t.h[1] = (bf16_t)f.y;
                  t.h[2] = (bf16_t)f.z; t.h[3] = (bf16_t)f.w;
            *(uint2*)&Bs[r][c4] = t.u;
        }
        if (k0 + 32 < K) {  // global_prefetch_b8 of next K tile
            __builtin_prefetch(&A[(size_t)(bm + (tid >> 1)) * K + k0 + 32], 0, 3);
            __builtin_prefetch(&W[(size_t)(bn + (tid & 63)) * K + k0 + 32], 0, 3);
        }
        __syncthreads();

        // ---- fragments: two ds_load_b128 each (CDNA5 16-bit layouts) ----
        Frag16 af[2], bf[2];
        #pragma unroll
        for (int mi = 0; mi < 2; ++mi) {
            int m = wm * 32 + mi * 16 + l16;
            af[mi].q[0] = *(const uint4*)&As[m][half * 8];        // K 0..7
            af[mi].q[1] = *(const uint4*)&As[m][16 + half * 8];   // K 16..23
        }
        #pragma unroll
        for (int ni = 0; ni < 2; ++ni) {
            int n = wn * 32 + ni * 16 + l16;
            bf[ni].q[0] = *(const uint4*)&Bs[n][half * 8];        // K 0..7
            bf[ni].q[1] = *(const uint4*)&Bs[n][16 + half * 8];   // K 16..23
        }
        #pragma unroll
        for (int mi = 0; mi < 2; ++mi)
            #pragma unroll
            for (int ni = 0; ni < 2; ++ni)
                acc[mi][ni] = __builtin_amdgcn_wmma_f32_16x16x32_bf16(
                    false, af[mi].v, false, bf[ni].v,
                    (short)0, acc[mi][ni], false, false);
        __syncthreads();
    }

    // ---- epilogue ----
    #pragma unroll
    for (int mi = 0; mi < 2; ++mi) {
        #pragma unroll
        for (int ni = 0; ni < 2; ++ni) {
            #pragma unroll
            for (int j = 0; j < 8; ++j) {
                int gr  = bm + wm * 32 + mi * 16 + j + half * 8;
                int col = bn + wn * 32 + ni * 16 + l16;
                if (gr >= M) continue;
                int b   = gr >> rpb_shift;
                int rib = gr & ((1 << rpb_shift) - 1);
                float v = acc[mi][ni][j];
                if (EPI == 0) {
                    float* C = (float*)Cout;
                    C[(size_t)(b * out_batch_stride + out_row_offset + rib) * N + col] = v;
                } else {
                    bf16_t* C = (bf16_t*)Cout;
                    int h = col >> 6, d = col & 63;
                    size_t idx = ((size_t)(b * HEADS + h) * kv_total +
                                  (out_row_offset + rib)) * HEAD_DIM + d;
                    C[idx] = (bf16_t)v;
                }
            }
        }
    }
}

// ---------------------------------------------------------------------------
// RMS-norm (+ interleaved RoPE on first 8 heads, pos < rope_limit)
// One wave32 per (b,h,pos) row of 64; lane owns pair (2l, 2l+1).
// src: f32 [b, rows, 1024] (col = h*64+d) ; dst: bf16 [b, h, rows, 64]
// ---------------------------------------------------------------------------
__global__ __launch_bounds__(256)
void norm_rope_kernel(const float* __restrict__ src, bf16_t* __restrict__ dst,
                      const float* __restrict__ normw,
                      const float* __restrict__ fcos, const float* __restrict__ fsin,
                      int rows, int rope_limit)
{
    const int wgl  = blockIdx.x * 8 + (threadIdx.x >> 5);
    const int lane = threadIdx.x & 31;
    if (wgl >= BATCH * HEADS * rows) return;
    const int b   = wgl / (HEADS * rows);
    const int rem = wgl - b * HEADS * rows;
    const int h   = rem / rows;
    const int pos = rem - h * rows;

    const float2 xv =
        *(const float2*)&src[((size_t)(b * rows + pos)) * DIM + h * HEAD_DIM + 2 * lane];
    float x0 = xv.x, x1 = xv.y;

    float ss = x0 * x0 + x1 * x1;
    #pragma unroll
    for (int off = 16; off > 0; off >>= 1) ss += __shfl_xor(ss, off, 32);
    const float r = rsqrtf(ss * (1.0f / HEAD_DIM) + 1e-6f);

    x0 *= r * normw[h * HEAD_DIM + 2 * lane];
    x1 *= r * normw[h * HEAD_DIM + 2 * lane + 1];

    if (h < HEADS / 2 && pos < rope_limit) {
        const float c  = fcos[pos * 32 + lane];
        const float sn = fsin[pos * 32 + lane];
        const float re = x0 * c - x1 * sn;
        const float im = x0 * sn + x1 * c;
        x0 = re; x1 = im;
    }
    const size_t o = ((size_t)(b * HEADS + h) * rows + pos) * HEAD_DIM + 2 * lane;
    dst[o]     = (bf16_t)x0;
    dst[o + 1] = (bf16_t)x1;
}

// ---------------------------------------------------------------------------
// Flash attention: one block (4 waves) per (b*h, 64-row q tile), kv tiles 64.
// Q/K tiles: async global->LDS b128. V staged transposed (Vt[d][kv]) so the
// P*V B-fragments are contiguous b128 loads. Online softmax in LDS.
// ---------------------------------------------------------------------------
__global__ __launch_bounds__(128)
void flash_attn_kernel(const bf16_t* __restrict__ Qb, const bf16_t* __restrict__ Kb,
                       const bf16_t* __restrict__ Vb,
                       const unsigned char* __restrict__ m_self,
                       const unsigned char* __restrict__ m_text,
                       const unsigned char* __restrict__ m_spk,
                       const unsigned char* __restrict__ m_cap,
                       bf16_t* __restrict__ Y)
{
    __shared__ bf16_t Qs[64][72];    // 144B rows: 16B aligned, conflict-free
    __shared__ bf16_t KPs[64][72];   // K tile; reused to hold P
    __shared__ bf16_t Vt[64][72];    // transposed V: row = d, col = kv
    __shared__ float  Ss[64][66];
    __shared__ float  mrow[64], lrow[64], crow[64], mbias[64];

    const int bh   = blockIdx.x;      // b*HEADS + h
    const int b    = bh >> 4;
    const int qt   = blockIdx.y;
    const int tid  = threadIdx.x;
    const int lane = tid & 31;
    const int wave = tid >> 5;        // 0..3
    const int half = lane >> 4;
    const int l16  = lane & 15;

    const uint32_t qs_base  = (uint32_t)(uintptr_t)(&Qs[0][0]);
    const uint32_t kps_base = (uint32_t)(uintptr_t)(&KPs[0][0]);

    // ---- async-load Q tile (64x64 bf16 = 512 b128 chunks) ----
    const size_t qbase = ((size_t)bh * SEQ + qt * 64) * HEAD_DIM;
    #pragma unroll
    for (int i = 0; i < 4; ++i) {
        int e = tid + i * 128;
        int r = e >> 3, c8 = (e & 7) * 8;
        ASYNC_G2L_B128(qs_base + (uint32_t)(r * 144 + c8 * 2),
                       &Qb[qbase + (size_t)r * HEAD_DIM + c8]);
    }
    if (tid < 64) { mrow[tid] = -3.0e38f; lrow[tid] = 0.0f; }
    WAIT_ASYNCCNT0();
    __syncthreads();

    // Q fragments for this wave's 16 rows (two K-steps over d)
    Frag16 qf[2];
    #pragma unroll
    for (int ks = 0; ks < 2; ++ks) {
        const int m = wave * 16 + l16;
        qf[ks].q[0] = *(const uint4*)&Qs[m][ks * 32 + half * 8];
        qf[ks].q[1] = *(const uint4*)&Qs[m][ks * 32 + 16 + half * 8];
    }

    v8f yacc[4];
    #pragma unroll
    for (int i = 0; i < 4; ++i) yacc[i] = v8f_zero();

    const size_t kvbase = (size_t)bh * KV_TOTAL * HEAD_DIM;

    for (int kt0 = 0; kt0 < KV_TOTAL; kt0 += 64) {
        __syncthreads();  // previous P/V consumers done before overwrite

        // K tile: async global -> LDS (row-major)
        #pragma unroll
        for (int i = 0; i < 4; ++i) {
            int e = tid + i * 128;
            int r = e >> 3, c8 = (e & 7) * 8;
            ASYNC_G2L_B128(kps_base + (uint32_t)(r * 144 + c8 * 2),
                           &Kb[kvbase + (size_t)(kt0 + r) * HEAD_DIM + c8]);
        }
        // V tile: b128 global load, transposed LDS store
        #pragma unroll
        for (int i = 0; i < 4; ++i) {
            int e = tid + i * 128;
            int r = e >> 3, c8 = (e & 7) * 8;
            H8 u; u.q = *(const uint4*)&Vb[kvbase + (size_t)(kt0 + r) * HEAD_DIM + c8];
            #pragma unroll
            for (int j = 0; j < 8; ++j) Vt[c8 + j][r] = u.h[j];
        }
        if (tid < 64) {
            const int kv = kt0 + tid;
            unsigned char mk;
            if      (kv < SEQ)             mk = m_self[b * SEQ + kv];
            else if (kv < SEQ + 512)       mk = m_text[b * 512 + (kv - SEQ)];
            else if (kv < SEQ + 512 + 64)  mk = m_spk [b * 64  + (kv - SEQ - 512)];
            else                           mk = m_cap [b * 256 + (kv - SEQ - 576)];
            mbias[tid] = mk ? 0.0f : -1.0e30f;
        }
        WAIT_ASYNCCNT0();
        __syncthreads();

        // ---- S = scale * Q K^T ----
        #pragma unroll
        for (int ns = 0; ns < 4; ++ns) {
            v8f s = v8f_zero();
            #pragma unroll
            for (int ks = 0; ks < 2; ++ks) {
                Frag16 kf;
                const int kcol = ns * 16 + l16;       // B column = key index
                kf.q[0] = *(const uint4*)&KPs[kcol][ks * 32 + half * 8];
                kf.q[1] = *(const uint4*)&KPs[kcol][ks * 32 + 16 + half * 8];
                s = __builtin_amdgcn_wmma_f32_16x16x32_bf16(
                    false, qf[ks].v, false, kf.v, (short)0, s, false, false);
            }
            #pragma unroll
            for (int j = 0; j < 8; ++j)
                Ss[wave * 16 + j + half * 8][ns * 16 + l16] = s[j] * 0.125f;
        }
        __syncthreads();

        // ---- online softmax (threads 0..63, one row each) ----
        if (tid < 64) {
            const int r = tid;
            const float mprev = mrow[r];
            float mx = mprev;
            for (int c = 0; c < 64; ++c) mx = fmaxf(mx, Ss[r][c] + mbias[c]);
            const float corr = __expf(mprev - mx);
            float sum = 0.0f;
            for (int c = 0; c < 64; ++c) {
                const float p = __expf(Ss[r][c] + mbias[c] - mx);
                sum += p;
                KPs[r][c] = (bf16_t)p;    // P overwrites K tile
            }
            mrow[r] = mx;
            lrow[r] = lrow[r] * corr + sum;
            crow[r] = corr;
        }
        __syncthreads();

        // ---- rescale accumulators, then Y += P V ----
        #pragma unroll
        for (int ns = 0; ns < 4; ++ns)
            #pragma unroll
            for (int j = 0; j < 8; ++j)
                yacc[ns][j] *= crow[wave * 16 + j + half * 8];

        #pragma unroll
        for (int ks = 0; ks < 2; ++ks) {
            Frag16 pf;
            const int m = wave * 16 + l16;
            pf.q[0] = *(const uint4*)&KPs[m][ks * 32 + half * 8];
            pf.q[1] = *(const uint4*)&KPs[m][ks * 32 + 16 + half * 8];
            #pragma unroll
            for (int ns = 0; ns < 4; ++ns) {
                Frag16 vf;
                const int d = ns * 16 + l16;          // B column = d index
                vf.q[0] = *(const uint4*)&Vt[d][ks * 32 + half * 8];
                vf.q[1] = *(const uint4*)&Vt[d][ks * 32 + 16 + half * 8];
                yacc[ns] = __builtin_amdgcn_wmma_f32_16x16x32_bf16(
                    false, pf.v, false, vf.v, (short)0, yacc[ns], false, false);
            }
        }
    }
    __syncthreads();

    // ---- normalize and store y as bf16 [b, q, h*64 + d] ----
    const int h = bh & 15;
    #pragma unroll
    for (int ns = 0; ns < 4; ++ns) {
        #pragma unroll
        for (int j = 0; j < 8; ++j) {
            const int r = wave * 16 + j + half * 8;
            const float l = fmaxf(lrow[r], 1e-30f);
            const float v = yacc[ns][j] / l;
            const int q = qt * 64 + r;
            const size_t o = ((size_t)(b * SEQ + q)) * DIM + h * HEAD_DIM + ns * 16 + l16;
            Y[o] = (bf16_t)v;
        }
    }
}

// ---------------------------------------------------------------------------
// y <- y * sigmoid(g)   (in-place on the bf16 attention output)
// ---------------------------------------------------------------------------
__global__ __launch_bounds__(256)
void gate_kernel(bf16_t* __restrict__ Y, const float* __restrict__ G, size_t n)
{
    const size_t i = (size_t)blockIdx.x * blockDim.x + threadIdx.x;
    if (i < n) {
        const float g = G[i];
        const float y = (float)Y[i];
        Y[i] = (bf16_t)(y / (1.0f + __expf(-g)));
    }
}

// ---------------------------------------------------------------------------
extern "C" void kernel_launch(void* const* d_in, const int* in_sizes, int n_in,
                              void* d_out, int out_size, void* d_ws, size_t ws_size,
                              hipStream_t stream)
{
    const float* x    = (const float*)d_in[0];
    const float* txt  = (const float*)d_in[1];
    const float* spk  = (const float*)d_in[2];
    const float* cap  = (const float*)d_in[3];
    const unsigned char* m_self = (const unsigned char*)d_in[4];
    const unsigned char* m_text = (const unsigned char*)d_in[5];
    const unsigned char* m_spk  = (const unsigned char*)d_in[6];
    const unsigned char* m_cap  = (const unsigned char*)d_in[7];
    const float* fcos = (const float*)d_in[8];
    const float* fsin = (const float*)d_in[9];
    const float* wq      = (const float*)d_in[10];
    const float* wk      = (const float*)d_in[11];
    const float* wv      = (const float*)d_in[12];
    const float* wk_text = (const float*)d_in[13];
    const float* wv_text = (const float*)d_in[14];
    const float* wk_spk  = (const float*)d_in[15];
    const float* wv_spk  = (const float*)d_in[16];
    const float* wk_cap  = (const float*)d_in[17];
    const float* wv_cap  = (const float*)d_in[18];
    const float* gate_w  = (const float*)d_in[19];
    const float* wo      = (const float*)d_in[20];
    const float* qnw     = (const float*)d_in[21];
    const float* knw     = (const float*)d_in[22];

    char* wsB = (char*)d_ws;
    size_t off = 0;
    float*  Qf = (float*) (wsB + off); off += (size_t)BATCH * SEQ      * DIM * 4;
    float*  Kf = (float*) (wsB + off); off += (size_t)BATCH * KV_TOTAL * DIM * 4;
    float*  Gf = (float*) (wsB + off); off += (size_t)BATCH * SEQ      * DIM * 4;
    bf16_t* Qb = (bf16_t*)(wsB + off); off += (size_t)BATCH * SEQ      * DIM * 2;
    bf16_t* Kb = (bf16_t*)(wsB + off); off += (size_t)BATCH * KV_TOTAL * DIM * 2;
    bf16_t* Vb = (bf16_t*)(wsB + off); off += (size_t)BATCH * KV_TOTAL * DIM * 2;
    bf16_t* Yb = (bf16_t*)(wsB + off); off += (size_t)BATCH * SEQ      * DIM * 2;

    // ---- projections (bf16 WMMA, f32 accumulate) ----
    gemm_wmma_kernel<float, 0><<<dim3(64, 16), 256, 0, stream>>>(
        x, wq, Qf, 8192, 1024, 1024, 11, 2048, 0, 0);
    gemm_wmma_kernel<float, 0><<<dim3(64, 16), 256, 0, stream>>>(
        x, wk, Kf, 8192, 1024, 1024, 11, 2880, 0, 0);
    gemm_wmma_kernel<float, 0><<<dim3(16, 16), 256, 0, stream>>>(
        txt, wk_text, Kf, 2048, 1024, 1024, 9, 2880, 2048, 0);
    gemm_wmma_kernel<float, 0><<<dim3(2, 16), 256, 0, stream>>>(
        spk, wk_spk, Kf, 256, 1024, 512, 6, 2880, 2560, 0);
    gemm_wmma_kernel<float, 0><<<dim3(8, 16), 256, 0, stream>>>(
        cap, wk_cap, Kf, 1024, 1024, 768, 8, 2880, 2624, 0);
    // V projections go straight to bf16 [b,h,kv,d]
    gemm_wmma_kernel<float, 1><<<dim3(64, 16), 256, 0, stream>>>(
        x, wv, Vb, 8192, 1024, 1024, 11, 0, 0, KV_TOTAL);
    gemm_wmma_kernel<float, 1><<<dim3(16, 16), 256, 0, stream>>>(
        txt, wv_text, Vb, 2048, 1024, 1024, 9, 0, 2048, KV_TOTAL);
    gemm_wmma_kernel<float, 1><<<dim3(2, 16), 256, 0, stream>>>(
        spk, wv_spk, Vb, 256, 1024, 512, 6, 0, 2560, KV_TOTAL);
    gemm_wmma_kernel<float, 1><<<dim3(8, 16), 256, 0, stream>>>(
        cap, wv_cap, Vb, 1024, 1024, 768, 8, 0, 2624, KV_TOTAL);
    // gate logits
    gemm_wmma_kernel<float, 0><<<dim3(64, 16), 256, 0, stream>>>(
        x, gate_w, Gf, 8192, 1024, 1024, 11, 2048, 0, 0);

    // ---- RMS norm + RoPE into bf16 attention layouts ----
    norm_rope_kernel<<<dim3((BATCH * HEADS * SEQ) / 8), 256, 0, stream>>>(
        Qf, Qb, qnw, fcos, fsin, SEQ, SEQ);
    norm_rope_kernel<<<dim3((BATCH * HEADS * KV_TOTAL) / 8), 256, 0, stream>>>(
        Kf, Kb, knw, fcos, fsin, KV_TOTAL, SEQ);

    // ---- flash attention ----
    flash_attn_kernel<<<dim3(BATCH * HEADS, SEQ / 64), 128, 0, stream>>>(
        Qb, Kb, Vb, m_self, m_text, m_spk, m_cap, Yb);

    // ---- gating ----
    const size_t ntot = (size_t)BATCH * SEQ * DIM;
    gate_kernel<<<dim3((unsigned)(ntot / 256)), 256, 0, stream>>>(Yb, Gf, ntot);

    // ---- output projection ----
    gemm_wmma_kernel<bf16_t, 0><<<dim3(64, 16), 256, 0, stream>>>(
        Yb, wo, (float*)d_out, 8192, 1024, 1024, 11, 2048, 0, 0);
}
